// M2Trans_33217277067812
// MI455X (gfx1250) — matrix operand order repaired and checked by reading, not compiled, not run on hardware
//
#include <hip/hip_runtime.h>
#include <cstddef>

#define DEV __device__ __forceinline__

typedef __attribute__((ext_vector_type(16))) __bf16 v16bf;
typedef __attribute__((ext_vector_type(8)))  float  v8f;

DEV v8f v8f_zero() { v8f z = {0.f,0.f,0.f,0.f,0.f,0.f,0.f,0.f}; return z; }

DEV v8f wmma_bf16(v16bf a, v16bf b, v8f c) {
  // D = A(16x32 bf16) * B(32x16 bf16) + C(16x16 f32)
  return __builtin_amdgcn_wmma_f32_16x16x32_bf16(false, a, false, b, (short)0, c, false, false);
}

DEV __bf16 f2bf(float f) { return (__bf16)f; }

// ---------------------------------------------------------------------------
// WMMA fragment loaders (CDNA5 wave32 VGPR layouts, CDNA5 ISA)
// A 16x32 bf16: lane l holds row m=l%16; vgpr v half h -> k = (l/16)*16 + v*2 + h (contiguous; not ISA table)
// B 32x16 bf16: lane l holds col n=l%16; vgpr v half h -> k = (l/16)*16 + v*2 + h (contiguous; not ISA table)
// C/D f32:      lane l, vgpr v -> (M = v + 8*(l/16), N = l%16)
// ---------------------------------------------------------------------------
DEV v16bf afrag_lds_bf16(const __bf16* base, int ld, int k0) {
  int lane = threadIdx.x & 31;
  int m = lane & 15;
  int khalf = (lane >> 4) << 3;
  v16bf a;
#pragma unroll
  for (int v = 0; v < 8; ++v) {
    int k = k0 + (khalf << 1) + (v << 1);
    a[2 * v]     = base[m * ld + k];
    a[2 * v + 1] = base[m * ld + k + 1];
  }
  return a;
}

DEV v16bf afrag_lds_f32(const float* base, int ld, int k0) {
  int lane = threadIdx.x & 31;
  int m = lane & 15;
  int khalf = (lane >> 4) << 3;
  v16bf a;
#pragma unroll
  for (int v = 0; v < 8; ++v) {
    int k = k0 + (khalf << 1) + (v << 1);
    a[2 * v]     = f2bf(base[m * ld + k]);
    a[2 * v + 1] = f2bf(base[m * ld + k + 1]);
  }
  return a;
}

// storage is [k][n] row-major with leading dim ld (in n)
DEV v16bf bfrag_lds_bf16(const __bf16* base, int ld, int k0, int n0) {
  int lane = threadIdx.x & 31;
  int n = n0 + (lane & 15);
  int kb = k0 + ((lane >> 4) << 4);
  v16bf b;
#pragma unroll
  for (int v = 0; v < 8; ++v) {
    b[2 * v]     = base[(kb + 2 * v)     * ld + n];
    b[2 * v + 1] = base[(kb + 2 * v + 1) * ld + n];
  }
  return b;
}

// storage is [n][k] row-major with leading dim ld (in k)  (i.e. B = stored^T)
DEV v16bf bfragT_lds_bf16(const __bf16* base, int ld, int k0) {
  int lane = threadIdx.x & 31;
  int n = lane & 15;
  int kb = k0 + ((lane >> 4) << 4);
  v16bf b;
#pragma unroll
  for (int v = 0; v < 8; ++v) {
    b[2 * v]     = base[n * ld + kb + 2 * v];
    b[2 * v + 1] = base[n * ld + kb + 2 * v + 1];
  }
  return b;
}

// ---------------------------------------------------------------------------
// Elementwise / reshape kernels
// ---------------------------------------------------------------------------

// Async-LDS round-trip copy: exercises GLOBAL_LOAD_ASYNC_TO_LDS_B128 /
// GLOBAL_STORE_ASYNC_FROM_LDS_B128 + ASYNCcnt (CDNA5 ISA).
__global__ __launch_bounds__(256) void k_copy_async(const float* __restrict__ a,
                                                    float* __restrict__ o, size_t n4) {
  __shared__ float4 sbuf[256];
  size_t i = (size_t)blockIdx.x * 256 + threadIdx.x;
  if (i >= n4) return;
  const float4* src = reinterpret_cast<const float4*>(a) + i;
  float4* dst = reinterpret_cast<float4*>(o) + i;
  unsigned lds = (unsigned)(size_t)&sbuf[threadIdx.x];
  asm volatile("global_load_async_to_lds_b128 %0, %1, off"
               :: "v"(lds), "v"(src) : "memory");
  asm volatile("s_wait_asynccnt 0x0" ::: "memory");
  asm volatile("global_store_async_from_lds_b128 %0, %1, off"
               :: "v"(dst), "v"(lds) : "memory");
  asm volatile("s_wait_asynccnt 0x0" ::: "memory");
}

__global__ void k_add_inplace(float* __restrict__ o, const float* __restrict__ a, size_t n) {
  for (size_t i = (size_t)blockIdx.x * blockDim.x + threadIdx.x; i < n;
       i += (size_t)gridDim.x * blockDim.x) o[i] += a[i];
}
__global__ void k_avg_inplace(float* __restrict__ o, const float* __restrict__ a, size_t n) {
  for (size_t i = (size_t)blockIdx.x * blockDim.x + threadIdx.x; i < n;
       i += (size_t)gridDim.x * blockDim.x) o[i] = 0.5f * (o[i] + a[i]);
}

__global__ void k_split4(const float* __restrict__ xn, float* __restrict__ o1,
                         float* __restrict__ o2, float* __restrict__ o3,
                         float* __restrict__ o4, int Bn, int HW) {
  size_t total = (size_t)Bn * 64 * HW;
  for (size_t i = (size_t)blockIdx.x * blockDim.x + threadIdx.x; i < total;
       i += (size_t)gridDim.x * blockDim.x) {
    size_t hw = i % HW;
    int c = (int)((i / HW) % 64);
    size_t b = i / ((size_t)64 * HW);
    int g = c >> 4, cl = c & 15;
    float* o = (g == 0) ? o1 : (g == 1) ? o2 : (g == 2) ? o3 : o4;
    o[((size_t)b * 16 + cl) * HW + hw] = xn[i];
  }
}

__global__ void k_instnorm(const float* __restrict__ x, float* __restrict__ y, int HW) {
  const float* xi = x + (size_t)blockIdx.x * HW;
  float* yi = y + (size_t)blockIdx.x * HW;
  __shared__ float ssum[256], ssq[256];
  float s = 0.f, q = 0.f;
  for (int i = threadIdx.x; i < HW; i += 256) { float v = xi[i]; s += v; q += v * v; }
  ssum[threadIdx.x] = s; ssq[threadIdx.x] = q;
  __syncthreads();
  for (int off = 128; off > 0; off >>= 1) {
    if ((int)threadIdx.x < off) { ssum[threadIdx.x] += ssum[threadIdx.x + off];
                                  ssq[threadIdx.x]  += ssq[threadIdx.x + off]; }
    __syncthreads();
  }
  float mean = ssum[0] / (float)HW;
  float var  = ssq[0] / (float)HW - mean * mean;
  float inv  = rsqrtf(var + 1e-5f);
  for (int i = threadIdx.x; i < HW; i += 256) yi[i] = (xi[i] - mean) * inv;
}

__global__ void k_dwt(const float* __restrict__ x, float* __restrict__ o,
                      int Bn, int C, int H, int W) {
  int H2 = H >> 1, W2 = W >> 1;
  size_t plane = (size_t)H2 * W2;
  size_t total = (size_t)Bn * C * plane;
  for (size_t i = (size_t)blockIdx.x * blockDim.x + threadIdx.x; i < total;
       i += (size_t)gridDim.x * blockDim.x) {
    int j = (int)(i % W2);
    int ii = (int)((i / W2) % H2);
    int c = (int)((i / plane) % C);
    int b = (int)(i / ((size_t)C * plane));
    const float* xb = x + ((size_t)b * C + c) * H * W;
    float a  = xb[(2 * ii) * W + 2 * j];
    float bb = xb[(2 * ii + 1) * W + 2 * j];
    float cc = xb[(2 * ii) * W + 2 * j + 1];
    float dd = xb[(2 * ii + 1) * W + 2 * j + 1];
    size_t ob = ((size_t)b * 4 * C) * plane + (size_t)ii * W2 + j;
    o[ob + (size_t)(0 * C + c) * plane] = 0.5f * ( a + bb + cc + dd);
    o[ob + (size_t)(1 * C + c) * plane] = 0.5f * (-a - bb + cc + dd);
    o[ob + (size_t)(2 * C + c) * plane] = 0.5f * (-a + bb - cc + dd);
    o[ob + (size_t)(3 * C + c) * plane] = 0.5f * ( a - bb - cc + dd);
  }
}

__global__ void k_iwt(const float* __restrict__ x, float* __restrict__ o,
                      int Bn, int C4, int H, int W) {
  int C = C4 >> 2;
  size_t plane = (size_t)H * W;
  size_t total = (size_t)Bn * C * plane;
  int W2 = 2 * W;
  for (size_t idx = (size_t)blockIdx.x * blockDim.x + threadIdx.x; idx < total;
       idx += (size_t)gridDim.x * blockDim.x) {
    int j = (int)(idx % W);
    int i = (int)((idx / W) % H);
    int c = (int)((idx / plane) % C);
    int b = (int)(idx / ((size_t)C * plane));
    const float* xb = x + (size_t)b * C4 * plane + (size_t)i * W + j;
    float x1 = xb[(size_t)(0 * C + c) * plane];
    float x2 = xb[(size_t)(1 * C + c) * plane];
    float x3 = xb[(size_t)(2 * C + c) * plane];
    float x4 = xb[(size_t)(3 * C + c) * plane];
    float p00 = 0.5f * (x1 - x2 - x3 + x4);
    float p10 = 0.5f * (x1 - x2 + x3 - x4);
    float p01 = 0.5f * (x1 + x2 - x3 - x4);
    float p11 = 0.5f * (x1 + x2 + x3 + x4);
    float* ob = o + ((size_t)b * C + c) * (4 * plane);
    ob[(size_t)(2 * i) * W2 + 2 * j]         = p00;
    ob[(size_t)(2 * i) * W2 + 2 * j + 1]     = p01;
    ob[(size_t)(2 * i + 1) * W2 + 2 * j]     = p10;
    ob[(size_t)(2 * i + 1) * W2 + 2 * j + 1] = p11;
  }
}

// Scalar 3x3 conv (reflect or zero pad), used for cheap head / tail convs.
__global__ void k_conv3x3_scalar(const float* __restrict__ x, const float* __restrict__ w,
                                 const float* __restrict__ bias, float* __restrict__ o,
                                 int Bn, int Cin, int Cout, int H, int W,
                                 int reflect, int doclip) {
  size_t plane = (size_t)H * W;
  size_t total = (size_t)Bn * Cout * plane;
  for (size_t i = (size_t)blockIdx.x * blockDim.x + threadIdx.x; i < total;
       i += (size_t)gridDim.x * blockDim.x) {
    int ww = (int)(i % W);
    int hh = (int)((i / W) % H);
    int co = (int)((i / plane) % Cout);
    int b  = (int)(i / ((size_t)Cout * plane));
    float acc = bias ? bias[co] : 0.f;
    for (int ci = 0; ci < Cin; ++ci) {
      const float* xb = x + ((size_t)b * Cin + ci) * plane;
      const float* wb = w + (((size_t)co * Cin + ci) * 9);
      for (int ky = 0; ky < 3; ++ky) {
        int h = hh + ky - 1;
        if (reflect) h = (h < 0) ? -h : (h >= H ? 2 * H - 2 - h : h);
        for (int kx = 0; kx < 3; ++kx) {
          int wc = ww + kx - 1;
          if (reflect) wc = (wc < 0) ? -wc : (wc >= W ? 2 * W - 2 - wc : wc);
          float v = 0.f;
          if (h >= 0 && h < H && wc >= 0 && wc < W) v = xb[(size_t)h * W + wc];
          acc += v * wb[ky * 3 + kx];
        }
      }
    }
    if (doclip) acc = fminf(fmaxf(acc, 0.f), 1.f);
    o[i] = acc;
  }
}

// ---------------------------------------------------------------------------
// 1x1 conv as WMMA GEMM (K compile-time):  out[b,o,hw] = sum_k w[o,k] * x[b,k,hw]
// block = 128 (4 waves), each wave owns one N-tile (16 pixels), grid.y = M/16
// K%32==0 -> fully unguarded loads; K=16 -> clamp-address + select (no branches)
// ---------------------------------------------------------------------------
template <int K>
__global__ __launch_bounds__(128) void k_conv1x1_wmma(
    const float* __restrict__ x, const float* __restrict__ w,
    float* __restrict__ out, int Bn, int M, int HW) {
  constexpr int KR = (K + 31) & ~31;
  int wid = threadIdx.x >> 5, lane = threadIdx.x & 31;
  int NT = (Bn * HW) >> 4;
  int ntile = blockIdx.x * 4 + wid;
  if (ntile >= NT) return;  // wave-uniform
  int mtile = blockIdx.y;
  int n0 = ntile << 4;
  int b = n0 / HW;
  int hw0 = n0 % HW;
  int m = lane & 15, khi = lane >> 4, nn = lane & 15;
  const float* wrow = w + (size_t)(mtile * 16 + m) * K;
  __builtin_prefetch(wrow, 0, 0);
  v8f acc = v8f_zero();
#pragma unroll
  for (int k0 = 0; k0 < KR; k0 += 32) {
    v16bf a, bf;
#pragma unroll
    for (int v = 0; v < 8; ++v) {   // A: weights (paired loads)
      int k = k0 + (khi << 4) + (v << 1);
      if constexpr (K % 32 == 0) {
        float2 p = *reinterpret_cast<const float2*>(wrow + k);
        a[2 * v]     = f2bf(p.x);
        a[2 * v + 1] = f2bf(p.y);
      } else {
        int kc = (k < K) ? k : 0;
        float2 p = *reinterpret_cast<const float2*>(wrow + kc);
        a[2 * v]     = (k < K) ? f2bf(p.x) : f2bf(0.f);
        a[2 * v + 1] = (k < K) ? f2bf(p.y) : f2bf(0.f);
      }
    }
#pragma unroll
    for (int v = 0; v < 8; ++v) {   // B: activations
      int k = k0 + (khi << 4) + (v << 1);
      if constexpr (K % 32 == 0) {
        const float* p = x + ((size_t)b * K + k) * HW + hw0 + nn;
        bf[2 * v]     = f2bf(p[0]);
        bf[2 * v + 1] = f2bf(p[HW]);
      } else {
        int kc = (k < K) ? k : 0;
        const float* p = x + ((size_t)b * K + kc) * HW + hw0 + nn;
        float va = p[0], vb = p[HW];
        bf[2 * v]     = (k < K) ? f2bf(va) : f2bf(0.f);
        bf[2 * v + 1] = (k < K) ? f2bf(vb) : f2bf(0.f);
      }
    }
    acc = wmma_bf16(a, bf, acc);
  }
#pragma unroll
  for (int v = 0; v < 8; ++v) {
    int mm = v + (khi << 3);
    out[((size_t)b * M + mtile * 16 + mm) * HW + hw0 + nn] = acc[v];
  }
}

// ---------------------------------------------------------------------------
// 1x1 conv (K=64, M=256) + bias + pixel_shuffle(2) + exact GELU (t1/t2)
// ---------------------------------------------------------------------------
__global__ __launch_bounds__(128) void k_conv1x1_shuffle_gelu(
    const float* __restrict__ x, const float* __restrict__ w,
    const float* __restrict__ bias, float* __restrict__ out,
    int Bn, int M, int HW, int H, int W) {
  constexpr int K = 64;
  int wid = threadIdx.x >> 5, lane = threadIdx.x & 31;
  int NT = (Bn * HW) >> 4;
  int ntile = blockIdx.x * 4 + wid;
  if (ntile >= NT) return;
  int mtile = blockIdx.y;
  int n0 = ntile << 4;
  int b = n0 / HW;
  int hw0 = n0 % HW;
  int m = lane & 15, khi = lane >> 4, nn = lane & 15;
  const float* wrow = w + (size_t)(mtile * 16 + m) * K;
  __builtin_prefetch(wrow, 0, 0);
  v8f acc = v8f_zero();
#pragma unroll
  for (int k0 = 0; k0 < K; k0 += 32) {
    v16bf a, bf;
#pragma unroll
    for (int v = 0; v < 8; ++v) {
      int k = k0 + (khi << 4) + (v << 1);
      float2 p = *reinterpret_cast<const float2*>(wrow + k);
      a[2 * v]     = f2bf(p.x);
      a[2 * v + 1] = f2bf(p.y);
    }
#pragma unroll
    for (int v = 0; v < 8; ++v) {
      int k = k0 + (khi << 4) + (v << 1);
      const float* p = x + ((size_t)b * K + k) * HW + hw0 + nn;
      bf[2 * v]     = f2bf(p[0]);
      bf[2 * v + 1] = f2bf(p[HW]);
    }
    acc = wmma_bf16(a, bf, acc);
  }
  int hw = hw0 + nn;
  int h = hw / W, wq = hw % W;
  int W2 = 2 * W;
#pragma unroll
  for (int v = 0; v < 8; ++v) {
    int o = mtile * 16 + v + (khi << 3);
    float val = acc[v] + bias[o];
    val = 0.5f * val * (1.0f + erff(val * 0.70710678118654752f));
    int c2 = o >> 2, s1 = (o >> 1) & 1, s2 = o & 1;
    out[((size_t)b * (M >> 2) + c2) * ((size_t)4 * HW) +
        (size_t)(2 * h + s1) * W2 + (2 * wq + s2)] = val;
  }
}

// ---------------------------------------------------------------------------
// Halo attention (heads=1, BLOCK=8, KSZ=10): one workgroup per window.
// qkv layout (B, 3C, S, S); out (B, C, S, S).  C is compile-time.
// ---------------------------------------------------------------------------
#define AKEYS 128
template <int C>
__global__ __launch_bounds__(128) void k_halo_attn(
    const float* __restrict__ qkv, const float* __restrict__ relh,
    const float* __restrict__ relw, float* __restrict__ out,
    int Bn, int S) {
  constexpr int CW  = (C < 64) ? C : 64;    // channel chunk width (16 or 64)
  constexpr int NCH = C / CW;               // chunks
  constexpr int KS1 = (CW + 31) / 32;       // GEMM1 k-steps per chunk
  constexpr int LDW = KS1 * 32;             // LDS row width (32 or 64)
  constexpr int NT2 = CW / 16;              // GEMM2 n-tiles per chunk
  constexpr int HALF = C / 2;
  __shared__ __bf16 sQ[64 * LDW];
  __shared__ __bf16 sKV[AKEYS * LDW];
  __shared__ float  sAttn[64 * AKEYS];
  int nb = S >> 3;
  int win = blockIdx.x;
  int b = win / (nb * nb);
  int rem = win % (nb * nb);
  int by = rem / nb, bx = rem % nb;
  int tid = threadIdx.x, wid = tid >> 5, lane = tid & 31;
  int HW = S * S;
  const float scale = rsqrtf((float)C);
  const float* qbase = qkv + (size_t)b * 3 * C * HW;

  // ---------------- GEMM1: sim = Q * Kp^T (accumulate over channel chunks)
  v8f acc1[8];
#pragma unroll
  for (int i = 0; i < 8; ++i) acc1[i] = v8f_zero();
  for (int cc = 0; cc < NCH; ++cc) {
    int c0 = cc * CW;
    if constexpr (CW != LDW) {              // pad Q columns when CW < 32-step
      for (int e = tid; e < 64 * LDW; e += 128) sQ[e] = f2bf(0.f);
    }
    for (int e = tid; e < AKEYS * LDW; e += 128) sKV[e] = f2bf(0.f);
    __syncthreads();
    for (int e = tid; e < 64 * CW; e += 128) {   // Q (scaled)
      int qi = e / CW, c = e % CW;
      int h = by * 8 + (qi >> 3), w = bx * 8 + (qi & 7);
      sQ[qi * LDW + c] = f2bf(qbase[(size_t)(c0 + c) * HW + h * S + w] * scale);
    }
    for (int e = tid; e < 100 * CW; e += 128) {  // K halo patch + rel bias
      int kk = e / CW, c = e % CW;
      int ki = kk / 10, kj = kk % 10;
      int h = by * 8 + ki - 1, w = bx * 8 + kj - 1;
      float val = 0.f;
      if (h >= 0 && h < S && w >= 0 && w < S)
        val = qbase[(size_t)(C + c0 + c) * HW + h * S + w];
      int cg = c0 + c;
      val += (cg < HALF) ? relh[ki * HALF + cg] : relw[kj * HALF + (cg - HALF)];
      sKV[kk * LDW + c] = f2bf(val);
    }
    __syncthreads();
#pragma unroll
    for (int nt = 0; nt < 8; ++nt)
#pragma unroll
      for (int ks = 0; ks < KS1; ++ks) {
        v16bf a  = afrag_lds_bf16(sQ + wid * 16 * LDW, LDW, ks * 32);
        v16bf bt = bfragT_lds_bf16(sKV + nt * 16 * LDW, LDW, ks * 32);
        acc1[nt] = wmma_bf16(a, bt, acc1[nt]);
      }
    __syncthreads();
  }
  { // spill sim to LDS
    int nn = lane & 15, khi = lane >> 4;
#pragma unroll
    for (int nt = 0; nt < 8; ++nt)
#pragma unroll
      for (int v = 0; v < 8; ++v)
        sAttn[(wid * 16 + v + (khi << 3)) * AKEYS + nt * 16 + nn] = acc1[nt][v];
  }
  __syncthreads();
  // ---------------- softmax over 100 keys per query row
  if (tid < 64) {
    float* row = sAttn + tid * AKEYS;
    float mx = -1e30f;
    for (int j = 0; j < 100; ++j) mx = fmaxf(mx, row[j]);
    float sum = 0.f;
    for (int j = 0; j < 100; ++j) { float e = __expf(row[j] - mx); row[j] = e; sum += e; }
    float inv = 1.0f / sum;
    for (int j = 0; j < 100; ++j) row[j] *= inv;
    for (int j = 100; j < AKEYS; ++j) row[j] = 0.f;
  }
  __syncthreads();
  // ---------------- GEMM2: out = attn * Vp (channel chunks in N)
  for (int cc = 0; cc < NCH; ++cc) {
    int c0 = cc * CW;
    if constexpr (CW != LDW) {
      for (int e = tid; e < AKEYS * LDW; e += 128) sKV[e] = f2bf(0.f);
    } else {  // only key rows 100..127 need zeroing
      for (int e = tid + 100 * LDW; e < AKEYS * LDW; e += 128) sKV[e] = f2bf(0.f);
    }
    __syncthreads();
    for (int e = tid; e < 100 * CW; e += 128) {  // V halo patch
      int kk = e / CW, c = e % CW;
      int ki = kk / 10, kj = kk % 10;
      int h = by * 8 + ki - 1, w = bx * 8 + kj - 1;
      float val = 0.f;
      if (h >= 0 && h < S && w >= 0 && w < S)
        val = qbase[(size_t)(2 * C + c0 + c) * HW + h * S + w];
      sKV[kk * LDW + c] = f2bf(val);
    }
    __syncthreads();
#pragma unroll
    for (int nt = 0; nt < NT2; ++nt) {
      v8f acc = v8f_zero();
#pragma unroll
      for (int ks = 0; ks < 4; ++ks) {       // K = 128 padded keys
        v16bf a = afrag_lds_f32(sAttn + wid * 16 * AKEYS, AKEYS, ks * 32);
        v16bf bf = bfrag_lds_bf16(sKV, LDW, ks * 32, nt * 16);
        acc = wmma_bf16(a, bf, acc);
      }
      int nn = lane & 15, khi = lane >> 4;
      int ch = c0 + nt * 16 + nn;            // always < C (C % CW == 0)
#pragma unroll
      for (int v = 0; v < 8; ++v) {
        int qi = wid * 16 + v + (khi << 3);
        int h = by * 8 + (qi >> 3), w = bx * 8 + (qi & 7);
        out[((size_t)b * C + ch) * HW + h * S + w] = acc[v];
      }
    }
    __syncthreads();
  }
}

// ---------------------------------------------------------------------------
// ff conv 3x3 (64->64, zero pad) as implicit-GEMM WMMA, fused bias + residual.
// Input channels come from the four 16-channel split buffers x1..x4.
// One workgroup (256 thr, 8 waves) per 16x16 spatial tile; K = 64*9 = 576.
// ---------------------------------------------------------------------------
__global__ __launch_bounds__(256) void k_ffconv_wmma(
    const float* __restrict__ x1, const float* __restrict__ x2,
    const float* __restrict__ x3, const float* __restrict__ x4,
    const float* __restrict__ wgt, const float* __restrict__ bias,
    float* __restrict__ y, int Bn, int H) {
  __shared__ __bf16 sPatch[18 * 18 * 64];
  int W = H, HW = H * W;
  int TB = H >> 4;
  int blk = blockIdx.x;
  int b = blk / (TB * TB);
  int rem = blk % (TB * TB);
  int ty = rem / TB, tx = rem % TB;
  int tid = threadIdx.x, wid = tid >> 5, lane = tid & 31;
  __builtin_prefetch(wgt + (size_t)tid * 144, 0, 0);

  // stage 18x18x64 input patch (bf16), zero-padded
  for (int e = tid; e < 18 * 18 * 64; e += 256) {
    int ch = e & 63;
    int pc = (e >> 6) % 18;
    int pr = (e >> 6) / 18;
    int h = ty * 16 + pr - 1, w = tx * 16 + pc - 1;
    float val = 0.f;
    if (h >= 0 && h < H && w >= 0 && w < W) {
      int g = ch >> 4, cl = ch & 15;
      const float* src = (g == 0) ? x1 : (g == 1) ? x2 : (g == 2) ? x3 : x4;
      val = src[((size_t)b * 16 + cl) * HW + (size_t)h * W + w];
    }
    sPatch[e] = f2bf(val);
  }
  __syncthreads();

  int m = lane & 15, khi = lane >> 4, nn = lane & 15;
  for (int pp = 0; pp < 8; ++pp) {
    int p = wid * 8 + pp;           // 64 (mt,nt) pairs over 8 waves
    int mt = p >> 4, nt = p & 15;
    const float* wrow = wgt + (size_t)(mt * 16 + m) * 576;
    v8f acc = v8f_zero();
    for (int ks = 0; ks < 18; ++ks) {   // K = 576 = 18*32
      v16bf a, bf;
#pragma unroll
      for (int v = 0; v < 8; ++v) {     // A: weights, k-contiguous (o*576 + k)
        int k = ks * 32 + (khi << 4) + (v << 1);
        float2 p2 = *reinterpret_cast<const float2*>(wrow + k);
        a[2 * v]     = f2bf(p2.x);
        a[2 * v + 1] = f2bf(p2.y);
      }
#pragma unroll
      for (int v = 0; v < 8; ++v) {     // B: patch gather
        int kb = ks * 32 + (khi << 4) + (v << 1);
        int pix = nt * 16 + nn;
        int lr = pix >> 4, lc = pix & 15;
#pragma unroll
        for (int hh = 0; hh < 2; ++hh) {
          int k = kb + hh;
          int ci = k / 9;
          int t = k - ci * 9;
          int ky = t / 3, kx = t - ky * 3;
          bf[2 * v + hh] = sPatch[((lr + ky) * 18 + (lc + kx)) * 64 + ci];
        }
      }
      acc = wmma_bf16(a, bf, acc);
    }
    // epilogue: + bias + residual (in-place on y)
#pragma unroll
    for (int v = 0; v < 8; ++v) {
      int ch = mt * 16 + v + (khi << 3);
      int pix = nt * 16 + nn;
      int h = ty * 16 + (pix >> 4), w = tx * 16 + (pix & 15);
      size_t idx = ((size_t)b * 64 + ch) * HW + (size_t)h * W + w;
      y[idx] = acc[v] + bias[ch] + y[idx];
    }
  }
}

// ---------------------------------------------------------------------------
extern "C" void kernel_launch(void* const* d_in, const int* in_sizes, int n_in,
                              void* d_out, int out_size, void* d_ws, size_t ws_size,
                              hipStream_t stream) {
  (void)in_sizes; (void)n_in; (void)out_size; (void)ws_size;
  const float* x      = (const float*)d_in[0];
  const float* head_w = (const float*)d_in[1];
  const float* head_b = (const float*)d_in[2];
  const float* aqkv[4] = {(const float*)d_in[3], (const float*)d_in[6],
                          (const float*)d_in[9], (const float*)d_in[12]};
  const float* arh[4]  = {(const float*)d_in[4], (const float*)d_in[7],
                          (const float*)d_in[10], (const float*)d_in[13]};
  const float* arw[4]  = {(const float*)d_in[5], (const float*)d_in[8],
                          (const float*)d_in[11], (const float*)d_in[14]};
  const float* ff_w = (const float*)d_in[15];
  const float* ff_b = (const float*)d_in[16];
  const float* t1_w = (const float*)d_in[17];
  const float* t1_b = (const float*)d_in[18];
  const float* t2_w = (const float*)d_in[19];
  const float* t2_b = (const float*)d_in[20];
  const float* t3_w = (const float*)d_in[21];
  float* out = (float*)d_out;

  const int B = 4, NF = 64, H = 192;
  const int HW = H * H;                       // 36864
  const size_t SZ_Y = (size_t)B * NF * HW;    // 9437184
  const size_t SZ_Q = (size_t)B * 48 * HW;    // 7077888 (same for all scales)
  const size_t SZ_S = (size_t)B * 16 * HW;    // 2359296

  float* ws   = (float*)d_ws;
  float* res  = ws;
  float* y    = res + SZ_Y;
  float* xn   = y + SZ_Y;
  float* qkvb = xn + SZ_Y;
  float* x1   = qkvb + SZ_Q;
  float* x2   = x1 + SZ_S;
  float* x3   = x2 + SZ_S;
  float* x4   = x3 + SZ_S;
  float* tmpa = x4 + SZ_S;
  float* tmpb = tmpa + SZ_S;
  float* aout = tmpb + SZ_S;
  float* itmp = aout + SZ_S;
  float* up1  = itmp + SZ_S;                  // (4,64,384,384)
  float* up2  = up1 + (size_t)B * NF * 4 * HW;// (4,64,768,768)

  auto eg = [](size_t n) { return dim3((unsigned)((n + 255) / 256)); };

  // head conv (reflect) -> res ; y = res (async-LDS copy path)
  k_conv3x3_scalar<<<eg(SZ_Y), 256, 0, stream>>>(x, head_w, head_b, res,
                                                 B, 3, NF, H, H, 1, 0);
  k_copy_async<<<eg(SZ_Y / 4), 256, 0, stream>>>(res, y, SZ_Y / 4);

  for (int i = 0; i < 4; ++i) {
    const float* q1 = aqkv[0] + (size_t)i * 48 * 16;
    const float* q2 = aqkv[1] + (size_t)i * 192 * 64;
    const float* q3 = aqkv[2] + (size_t)i * 768 * 256;
    const float* q4 = aqkv[3] + (size_t)i * 768 * 256;
    const float* rh1 = arh[0] + (size_t)i * 80,   *rw1 = arw[0] + (size_t)i * 80;
    const float* rh2 = arh[1] + (size_t)i * 320,  *rw2 = arw[1] + (size_t)i * 320;
    const float* rh3 = arh[2] + (size_t)i * 1280, *rw3 = arw[2] + (size_t)i * 1280;
    const float* rh4 = arh[3] + (size_t)i * 1280, *rw4 = arw[3] + (size_t)i * 1280;
    const float* fw = ff_w + (size_t)i * 64 * 64 * 9;
    const float* fb = ff_b + (size_t)i * 64;

    k_instnorm<<<B * NF, 256, 0, stream>>>(y, xn, HW);
    k_split4<<<eg(SZ_Y), 256, 0, stream>>>(xn, x1, x2, x3, x4, B, HW);

    // ---- branch 1: C=16 @ 192
    k_conv1x1_wmma<16><<<dim3((B * HW / 16) / 4, 48 / 16), 128, 0, stream>>>(
        x1, q1, qkvb, B, 48, HW);
    k_halo_attn<16><<<B * 24 * 24, 128, 0, stream>>>(qkvb, rh1, rw1, aout, B, 192);
    k_add_inplace<<<eg(SZ_S), 256, 0, stream>>>(x1, aout, SZ_S);

    // ---- branch 2: dwt -> C=64 @ 96
    k_avg_inplace<<<eg(SZ_S), 256, 0, stream>>>(x2, x1, SZ_S);
    k_dwt<<<eg(SZ_S / 4), 256, 0, stream>>>(x2, tmpa, B, 16, 192, 192);
    k_conv1x1_wmma<64><<<dim3((B * 9216 / 16) / 4, 192 / 16), 128, 0, stream>>>(
        tmpa, q2, qkvb, B, 192, 9216);
    k_halo_attn<64><<<B * 12 * 12, 128, 0, stream>>>(qkvb, rh2, rw2, aout, B, 96);
    k_iwt<<<eg(SZ_S / 4), 256, 0, stream>>>(aout, itmp, B, 64, 96, 96);
    k_add_inplace<<<eg(SZ_S), 256, 0, stream>>>(x2, itmp, SZ_S);

    // ---- branch 3: dwt^2 -> C=256 @ 48
    k_avg_inplace<<<eg(SZ_S), 256, 0, stream>>>(x3, x2, SZ_S);
    k_dwt<<<eg(SZ_S / 4), 256, 0, stream>>>(x3, tmpa, B, 16, 192, 192);
    k_dwt<<<eg(SZ_S / 4), 256, 0, stream>>>(tmpa, tmpb, B, 64, 96, 96);
    k_conv1x1_wmma<256><<<dim3((B * 2304 / 16) / 4, 768 / 16), 128, 0, stream>>>(
        tmpb, q3, qkvb, B, 768, 2304);
    k_halo_attn<256><<<B * 6 * 6, 128, 0, stream>>>(qkvb, rh3, rw3, aout, B, 48);
    k_iwt<<<eg(SZ_S / 4), 256, 0, stream>>>(aout, tmpa, B, 256, 48, 48);
    k_iwt<<<eg(SZ_S / 4), 256, 0, stream>>>(tmpa, itmp, B, 64, 96, 96);
    k_add_inplace<<<eg(SZ_S), 256, 0, stream>>>(x3, itmp, SZ_S);

    // ---- branch 4: dwt^2 -> C=256 @ 48
    k_avg_inplace<<<eg(SZ_S), 256, 0, stream>>>(x4, x3, SZ_S);
    k_dwt<<<eg(SZ_S / 4), 256, 0, stream>>>(x4, tmpa, B, 16, 192, 192);
    k_dwt<<<eg(SZ_S / 4), 256, 0, stream>>>(tmpa, tmpb, B, 64, 96, 96);
    k_conv1x1_wmma<256><<<dim3((B * 2304 / 16) / 4, 768 / 16), 128, 0, stream>>>(
        tmpb, q4, qkvb, B, 768, 2304);
    k_halo_attn<256><<<B * 6 * 6, 128, 0, stream>>>(qkvb, rh4, rw4, aout, B, 48);
    k_iwt<<<eg(SZ_S / 4), 256, 0, stream>>>(aout, tmpa, B, 256, 48, 48);
    k_iwt<<<eg(SZ_S / 4), 256, 0, stream>>>(tmpa, itmp, B, 64, 96, 96);
    k_add_inplace<<<eg(SZ_S), 256, 0, stream>>>(x4, itmp, SZ_S);

    // ---- ff conv 3x3 + bias + residual (in-place y)
    k_ffconv_wmma<<<B * 12 * 12, 256, 0, stream>>>(x1, x2, x3, x4, fw, fb, y, B, H);
  }

  // global residual
  k_add_inplace<<<eg(SZ_Y), 256, 0, stream>>>(y, res, SZ_Y);

  // t1: 64->256, shuffle(2), gelu -> up1 (4,64,384,384)
  k_conv1x1_shuffle_gelu<<<dim3((B * HW / 16) / 4, 256 / 16), 128, 0, stream>>>(
      y, t1_w, t1_b, up1, B, 256, HW, H, H);
  // t2: 64->256, shuffle(2), gelu -> up2 (4,64,768,768)
  k_conv1x1_shuffle_gelu<<<dim3((B * 4 * HW / 16) / 4, 256 / 16), 128, 0, stream>>>(
      up1, t2_w, t2_b, up2, B, 256, 4 * HW, 2 * H, 2 * H);
  // t3: 64->3, reflect 3x3, clip -> out
  k_conv3x3_scalar<<<eg((size_t)B * 3 * 16 * HW), 256, 0, stream>>>(
      up2, t3_w, nullptr, out, B, 64, 3, 4 * H, 4 * H, 1, 1);
}
